// GcnEncoderCell_29257317220432
// MI455X (gfx1250) — matrix operand order repaired and checked by reading, not compiled, not run on hardware
//
#include <hip/hip_runtime.h>
#include <stdint.h>
#include <math.h>

#define B_ 8
#define N_ 325
#define T_ 24
#define D_ 64
#define H_ 8
#define M_ 16
#define E2_ 128
#define DK_ 16
#define NEG_ 0.1f

typedef __attribute__((ext_vector_type(16))) __bf16 v16bf;
typedef __attribute__((ext_vector_type(8)))  float  v8f;
typedef __attribute__((ext_vector_type(4)))  float  v4f;
typedef __attribute__((ext_vector_type(4)))  unsigned int v4u;

union FragAB { v16bf v; uint32_t u[8]; v4u q[2]; };

#if defined(__has_builtin)
#if __has_builtin(__builtin_amdgcn_global_load_async_to_lds_b32)
#define HAVE_ASYNC_LDS 1
#endif
#if __has_builtin(__builtin_amdgcn_s_wait_asynccnt)
#define HAVE_WAIT_ASYNC 1
#endif
#endif

typedef __attribute__((address_space(1))) int g_int;
typedef __attribute__((address_space(3))) int l_int;

// native f32->bf16 (RTNE) so the compiler can emit v_cvt_pk_bf16_f32
__device__ __forceinline__ uint32_t pack_bf2(float a, float b) {
  union { __bf16 h[2]; uint32_t u; } r;
  r.h[0] = (__bf16)a; r.h[1] = (__bf16)b;
  return r.u;
}
__device__ __forceinline__ unsigned short bf16bits(float a) {
  union { __bf16 h; unsigned short s; } r;
  r.h = (__bf16)a;
  return r.s;
}
// 16-bit A-matrix 16x32 lane/VGPR layout (contiguous k order, matches B)
__device__ __forceinline__ int kkA(int j, int khalf) {
  return khalf * 16 + 2 * j;
}
// 16-bit B-matrix 32x16 lane/VGPR layout (contiguous k order, matches A)
__device__ __forceinline__ int kkB(int j, int khalf) {
  return khalf * 16 + 2 * j;
}
__device__ __forceinline__ float lrelu_(float x) { return x > 0.f ? x : NEG_ * x; }

// ---------------------------------------------------------------------------
// Kernel A: tfeat = tanh(mean_t(tXin[:,0]) @ tproj_w + b); emb = node_emb * tfeat
// ---------------------------------------------------------------------------
__global__ void __launch_bounds__(128) emb_kernel(
    const float* __restrict__ tXin, const float* __restrict__ node_emb,
    const float* __restrict__ tproj_w, const float* __restrict__ tproj_b,
    float* __restrict__ emb_ws) {
  const int b = blockIdx.x;
  const int tid = threadIdx.x;
  __shared__ float sMean[D_];
  __shared__ float sT[M_];
  if (tid < D_) {
    float s = 0.f;
    for (int t = 0; t < T_; ++t)
      s += tXin[((size_t)b * N_ * T_ + t) * D_ + tid];  // n = 0 slice
    sMean[tid] = s * (1.0f / (float)T_);
  }
  __syncthreads();
  if (tid < M_) {
    float a = tproj_b[tid];
    for (int d = 0; d < D_; ++d) a += sMean[d] * tproj_w[d * M_ + tid];
    sT[tid] = tanhf(a);
  }
  __syncthreads();
  for (int idx = tid; idx < N_ * M_; idx += 128) {
    int n = idx / M_, m = idx % M_;
    emb_ws[((size_t)b * N_ + n) * M_ + m] = node_emb[n * M_ + m] * sT[m];
  }
}

// ---------------------------------------------------------------------------
// Kernel B: support = matrix[b,t] @ hidden[b,:,t,:]; gcn_out = relu(support@W+b)
// one block = (b, t, 16-row n-tile); 4 waves, one 16-wide d-tile each
// ---------------------------------------------------------------------------
__global__ void __launch_bounds__(128) gcn_kernel(
    const float* __restrict__ hidden, const float* __restrict__ matrix,
    const float* __restrict__ gcn_w, const float* __restrict__ gcn_b,
    float* __restrict__ gcn_ws) {
  const int n0 = blockIdx.x * 16;
  const int t  = blockIdx.y;
  const int b  = blockIdx.z;
  const int tid = threadIdx.x;
  const int w = tid >> 5;
  const int lane = tid & 31;
  const int col = lane & 15;
  const int khalf = lane >> 4;
  const int d0 = w * 16;

  __shared__ __align__(64) float sMat[16 * 32];
  __shared__ __align__(64) float sHid[32 * D_];
  __shared__ __align__(64) float sSup[16 * D_];

  const float* matBase = matrix + (size_t)(b * T_ + t) * N_ * N_;
  v8f acc = {};
  for (int c = 0; c < 11; ++c) {         // K over m: 11 chunks of 32 (325 padded)
    const int m0 = c * 32;
    // A-tile must be exact-zero padded (kills the clamped garbage in B rows)
    for (int idx = tid; idx < 16 * 32; idx += 128) {
      const int nl = idx >> 5, ml = idx & 31;
      const int n = n0 + nl, m = m0 + ml;
      const int nc = n < N_ ? n : N_ - 1;
      const int mc = m < N_ ? m : N_ - 1;
      const float v = matBase[(size_t)nc * N_ + mc];
      sMat[idx] = (n < N_ && m < N_) ? v : 0.f;
    }
    // B-tile: clamped address, unconditional (padded rows are multiplied by 0)
#if defined(HAVE_ASYNC_LDS)
    for (int idx = tid; idx < 32 * D_; idx += 128) {
      const int ml = idx >> 6, d = idx & 63;
      int m = m0 + ml; if (m >= N_) m = N_ - 1;
      const float* g = hidden + ((size_t)(b * N_ + m) * T_ + t) * D_ + d;
      __builtin_amdgcn_global_load_async_to_lds_b32(
          (g_int*)(uintptr_t)g, (l_int*)(uintptr_t)&sHid[idx], 0, 0);
    }
#if defined(HAVE_WAIT_ASYNC)
    __builtin_amdgcn_s_wait_asynccnt(0);
#else
    asm volatile("s_wait_asynccnt 0" ::: "memory");
#endif
#else
    for (int idx = tid; idx < 32 * D_; idx += 128) {
      const int ml = idx >> 6, d = idx & 63;
      int m = m0 + ml; if (m >= N_) m = N_ - 1;
      sHid[idx] = hidden[((size_t)(b * N_ + m) * T_ + t) * D_ + d];
    }
#endif
    if (c + 1 < 11) {
      int mp = m0 + 32; if (mp >= N_) mp = N_ - 1;
      __builtin_prefetch(hidden + ((size_t)(b * N_ + mp) * T_ + t) * D_, 0, 0);
    }
    __syncthreads();
    FragAB fa, fb;
#pragma unroll
    for (int j = 0; j < 8; ++j) {
      const int ka = kkA(j, khalf);
      fa.u[j] = pack_bf2(sMat[col * 32 + ka], sMat[col * 32 + ka + 1]);
      const int kb = kkB(j, khalf);
      fb.u[j] = pack_bf2(sHid[kb * D_ + d0 + col], sHid[(kb + 1) * D_ + d0 + col]);
    }
    acc = __builtin_amdgcn_wmma_f32_16x16x32_bf16(false, fa.v, false, fb.v,
                                                  (short)0, acc, false, false);
    __syncthreads();
  }
#pragma unroll
  for (int r = 0; r < 8; ++r)
    sSup[(r + 8 * khalf) * D_ + d0 + col] = acc[r];
  __syncthreads();

  // gcn matmul: (16 x 64) @ (64 x 64); this wave owns d-tile d0
  v8f acc2 = {};
#pragma unroll
  for (int kc = 0; kc < 2; ++kc) {
    FragAB fa, fb;
#pragma unroll
    for (int j = 0; j < 8; ++j) {
      const int ka = kc * 32 + kkA(j, khalf);
      fa.u[j] = pack_bf2(sSup[col * D_ + ka], sSup[col * D_ + ka + 1]);
      const int kb = kc * 32 + kkB(j, khalf);
      fb.u[j] = pack_bf2(gcn_w[kb * D_ + d0 + col], gcn_w[(kb + 1) * D_ + d0 + col]);
    }
    acc2 = __builtin_amdgcn_wmma_f32_16x16x32_bf16(false, fa.v, false, fb.v,
                                                   (short)0, acc2, false, false);
  }
  const float bias = gcn_b[d0 + col];
#pragma unroll
  for (int r = 0; r < 8; ++r) {
    const int n = n0 + r + 8 * khalf;
    if (n < N_) {
      float v = acc2[r] + bias;
      gcn_ws[((size_t)(b * N_ + n) * T_ + t) * D_ + d0 + col] = fmaxf(v, 0.f);
    }
  }
}

// ---------------------------------------------------------------------------
// Kernel C: per-(b,n) fused attention cell. 256 threads = 8 waves.
// ---------------------------------------------------------------------------
#define SMEM_Q_OFF   98304            // 3 * 128*128 bf16 = 96 KB of weights
#define SMEM_K_OFF   (98304 + 12288)
#define SMEM_V_OFF   (98304 + 24576)
#define SMEM_EMB_OFF (98304 + 36864)  // 135168
#define SMEM_TOTAL   (135168 + 64)

__global__ void __launch_bounds__(256) attn_kernel(
    const float* __restrict__ hidden, const float* __restrict__ tXin,
    const float* __restrict__ WK, const float* __restrict__ WQ,
    const float* __restrict__ WV,
    const float* __restrict__ out_w, const float* __restrict__ out_b,
    const float* __restrict__ gate_w, const float* __restrict__ gate_b,
    const float* __restrict__ emb_ws, const float* __restrict__ gcn_ws,
    float* __restrict__ out) {
  __shared__ __align__(64) unsigned char smem[SMEM_TOTAL];
  unsigned short* sW = (unsigned short*)smem;       // [3][k_out][i] bf16
  float* sQ   = (float*)(smem + SMEM_Q_OFF);        // 24x128; later holds attn-out
  float* sK   = (float*)(smem + SMEM_K_OFF);
  float* sV   = (float*)(smem + SMEM_V_OFF);
  float* sVal = (float*)smem;                       // 24x64, aliases dead sW (phase>=3)
  float* sEmb = (float*)(smem + SMEM_EMB_OFF);

  const int bn = blockIdx.x;
  const int b = bn / N_, n = bn % N_;
  const int tid = threadIdx.x;
  const int w = tid >> 5, lane = tid & 31;
  const int col = lane & 15, khalf = lane >> 4;
  const size_t baseTD = (size_t)(b * N_ + n) * T_ * D_;

  if (tid < M_) sEmb[tid] = emb_ws[(size_t)bn * M_ + tid];
  __syncthreads();

  // Phase 1a: Wq/Wk/Wv = sum_m emb[m] * W[m] -> bf16 LDS [k_out][i]
  // vectorized 16B loads along contiguous k axis of W[m,i,k]
  {
    const float* Wsrc[3] = { WQ, WK, WV };
    float e[M_];
#pragma unroll
    for (int m = 0; m < M_; ++m) e[m] = sEmb[m];
    for (int p = 0; p < 3; ++p) {
      const float* Wp = Wsrc[p];
      unsigned short* sWp = sW + p * (128 * 128);
      for (int gg = 0; gg < 16; ++gg) {
        const int g = gg * 256 + tid;
        const int i = g >> 5;
        const int k4 = (g & 31) * 4;
        float a0 = 0.f, a1 = 0.f, a2 = 0.f, a3 = 0.f;
#pragma unroll
        for (int m = 0; m < M_; ++m) {
          const v4f wv = *(const v4f*)(Wp + ((size_t)m * E2_ + i) * E2_ + k4);
          a0 += e[m] * wv.x; a1 += e[m] * wv.y;
          a2 += e[m] * wv.z; a3 += e[m] * wv.w;
        }
        sWp[(k4 + 0) * 128 + i] = bf16bits(a0);
        sWp[(k4 + 1) * 128 + i] = bf16bits(a1);
        sWp[(k4 + 2) * 128 + i] = bf16bits(a2);
        sWp[(k4 + 3) * 128 + i] = bf16bits(a3);
      }
    }
  }
  __syncthreads();

  // Phase 1b: q/k/v = lrelu([hidden|tXin](24x128) @ W(128x128))
  // wave owns one t-tile (A fragments hoisted), 6 (p,ktile) output tiles
  const float* qkvA = hidden + baseTD;
  const float* qkvB = tXin + baseTD;
  {
    const int ttile = w & 1;
    const int trow = ttile * 16 + col;
    const int trc = trow < T_ ? trow : 0;   // clamp; garbage rows never stored
    FragAB af[4];
#pragma unroll
    for (int kc = 0; kc < 4; ++kc) {
      const float* src = (kc < 2) ? qkvA : qkvB;
      const int i0 = (kc & 1) * 32;
#pragma unroll
      for (int j = 0; j < 8; ++j) {
        const int ia = i0 + kkA(j, khalf);
        af[kc].u[j] = pack_bf2(src[trc * D_ + ia], src[trc * D_ + ia + 1]);
      }
    }
    for (int it = 0; it < 6; ++it) {
      const int idx6 = (w >> 1) + 4 * it;   // 0..23 over (p, ktile)
      const int p = idx6 >> 3;
      const int ktile = idx6 & 7;
      const unsigned short* Wrow = sW + p * (128 * 128) + (ktile * 16 + col) * 128;
      v8f acc = {};
#pragma unroll
      for (int kc = 0; kc < 4; ++kc) {
        FragAB fb;
        const v4u* bp = (const v4u*)(Wrow + kc * 32 + khalf * 16);
        fb.q[0] = bp[0]; fb.q[1] = bp[1];   // two ds_load_b128
        acc = __builtin_amdgcn_wmma_f32_16x16x32_bf16(false, af[kc].v, false, fb.v,
                                                      (short)0, acc, false, false);
      }
      float* dst = (p == 0) ? sQ : (p == 1) ? sK : sV;
#pragma unroll
      for (int r = 0; r < 8; ++r) {
        const int tr = ttile * 16 + r + 8 * khalf;
        if (tr < T_) dst[tr * E2_ + ktile * 16 + col] = lrelu_(acc[r]);
      }
    }
  }
  __syncthreads();

  // Phase 2: causal online-softmax attention per (h,t) row; result back into sQ
  if (tid < H_ * T_) {
    const int t = tid % T_;
    const int h = tid / T_;
    const int hk = h * DK_;
    float qr[DK_];
#pragma unroll
    for (int d = 0; d < DK_; ++d) qr[d] = sQ[t * E2_ + hk + d];
    float mx = -3.0e38f, den = 0.f;
    float o[DK_];
#pragma unroll
    for (int d = 0; d < DK_; ++d) o[d] = 0.f;
    for (int s = 0; s <= t; ++s) {
      float a = 0.f;
#pragma unroll
      for (int d = 0; d < DK_; ++d) a += qr[d] * sK[s * E2_ + hk + d];
      a *= 0.25f;                       // 1/sqrt(DK)
      const float nm = fmaxf(mx, a);
      const float fs = __expf(mx - nm);
      const float ee = __expf(a - nm);
      den = den * fs + ee;
#pragma unroll
      for (int d = 0; d < DK_; ++d)
        o[d] = o[d] * fs + ee * sV[s * E2_ + hk + d];
      mx = nm;
    }
    const float inv = 1.f / den;
#pragma unroll
    for (int d = 0; d < DK_; ++d) sQ[t * E2_ + hk + d] = o[d] * inv;
  }
  __syncthreads();

  // Phase 3: value = lrelu(attn_out(24x128) @ out_w(128x64) + out_b)
  {
    const int ttile = w >> 2, dt = w & 3;
    const int d0 = dt * 16;
    const int trow = ttile * 16 + col;
    const int trc = trow < T_ ? trow : 0;
    v8f acc = {};
    for (int kc = 0; kc < 4; ++kc) {
      FragAB fa, fb;
#pragma unroll
      for (int j = 0; j < 8; ++j) {
        const int ia = kc * 32 + kkA(j, khalf);
        fa.u[j] = pack_bf2(sQ[trc * E2_ + ia], sQ[trc * E2_ + ia + 1]);
        const int ib = kc * 32 + kkB(j, khalf);
        fb.u[j] = pack_bf2(out_w[ib * D_ + d0 + col], out_w[(ib + 1) * D_ + d0 + col]);
      }
      acc = __builtin_amdgcn_wmma_f32_16x16x32_bf16(false, fa.v, false, fb.v,
                                                    (short)0, acc, false, false);
    }
    const float bias = out_b[d0 + col];
#pragma unroll
    for (int r = 0; r < 8; ++r) {
      const int tr = ttile * 16 + r + 8 * khalf;
      if (tr < T_) sVal[tr * D_ + d0 + col] = lrelu_(acc[r] + bias);
    }
  }
  __syncthreads();

  // Phase 4: z = sigmoid([gcn|value](24x128) @ gate_w + gate_b); blend + residual
  {
    const int ttile = w >> 2, dt = w & 3;
    const int d0 = dt * 16;
    const int trow = ttile * 16 + col;
    const int trc = trow < T_ ? trow : 0;
    const float* gcnRow = gcn_ws + baseTD;
    v8f acc = {};
    for (int kc = 0; kc < 4; ++kc) {
      FragAB fa, fb;
      const int i0 = (kc & 1) * 32;
#pragma unroll
      for (int j = 0; j < 8; ++j) {
        const int ia = i0 + kkA(j, khalf);
        float f0, f1;
        if (kc < 2) { f0 = gcnRow[trc * D_ + ia]; f1 = gcnRow[trc * D_ + ia + 1]; }
        else        { f0 = sVal[trc * D_ + ia];   f1 = sVal[trc * D_ + ia + 1]; }
        fa.u[j] = pack_bf2(f0, f1);
        const int ib = kc * 32 + kkB(j, khalf);
        fb.u[j] = pack_bf2(gate_w[ib * D_ + d0 + col], gate_w[(ib + 1) * D_ + d0 + col]);
      }
      acc = __builtin_amdgcn_wmma_f32_16x16x32_bf16(false, fa.v, false, fb.v,
                                                    (short)0, acc, false, false);
    }
    const float bias = gate_b[d0 + col];
#pragma unroll
    for (int r = 0; r < 8; ++r) {
      const int tr = ttile * 16 + r + 8 * khalf;
      if (tr < T_) {
        const int d = d0 + col;
        const float z = 1.f / (1.f + __expf(-(acc[r] + bias)));
        const float g = gcnRow[tr * D_ + d];
        const float vv = sVal[tr * D_ + d];
        out[baseTD + tr * D_ + d] = z * g + (1.f - z) * vv + hidden[baseTD + tr * D_ + d];
      }
    }
  }
}

// ---------------------------------------------------------------------------
extern "C" void kernel_launch(void* const* d_in, const int* in_sizes, int n_in,
                              void* d_out, int out_size, void* d_ws, size_t ws_size,
                              hipStream_t stream) {
  (void)in_sizes; (void)n_in; (void)out_size; (void)ws_size;
  const float* hidden   = (const float*)d_in[0];
  const float* tXin     = (const float*)d_in[1];
  const float* matrix   = (const float*)d_in[2];
  const float* gcn_w    = (const float*)d_in[3];
  const float* gcn_b    = (const float*)d_in[4];
  const float* node_emb = (const float*)d_in[5];
  const float* tproj_w  = (const float*)d_in[6];
  const float* tproj_b  = (const float*)d_in[7];
  const float* WK       = (const float*)d_in[8];
  const float* WQ       = (const float*)d_in[9];
  const float* WV       = (const float*)d_in[10];
  const float* out_w    = (const float*)d_in[11];
  const float* out_b    = (const float*)d_in[12];
  const float* gate_w   = (const float*)d_in[13];
  const float* gate_b   = (const float*)d_in[14];
  float* out = (float*)d_out;

  float* emb_ws = (float*)d_ws;                                        // B*N*M
  float* gcn_ws = (float*)((char*)d_ws +
                           (size_t)B_ * N_ * M_ * sizeof(float));      // B*N*T*D

  emb_kernel<<<B_, 128, 0, stream>>>(tXin, node_emb, tproj_w, tproj_b, emb_ws);
  gcn_kernel<<<dim3((N_ + 15) / 16, T_, B_), 128, 0, stream>>>(
      hidden, matrix, gcn_w, gcn_b, gcn_ws);
  attn_kernel<<<B_ * N_, 256, 0, stream>>>(hidden, tXin, WK, WQ, WV, out_w, out_b,
                                           gate_w, gate_b, emb_ws, gcn_ws, out);
}
